// HeadVer4_14035953123649
// MI455X (gfx1250) — matrix operand order repaired and checked by reading, not compiled, not run on hardware
//
#include <hip/hip_runtime.h>

#define DEVINL __device__ __forceinline__

typedef __bf16 bf16;
typedef __attribute__((ext_vector_type(16))) __bf16 v16bf;
typedef __attribute__((ext_vector_type(8)))  __bf16 bf16x8;
typedef __attribute__((ext_vector_type(8)))  float  v8f;
typedef __attribute__((ext_vector_type(4)))  float  f32x4;

static constexpr int BB = 4, TT = 2048, CC = 1024;
static constexpr int MT = BB * TT;          // 8192 rows total
static constexpr int LDP = 40;              // LDS row stride in halves (32 + 8 pad)

// ---------------- fp32 -> bf16 convert ----------------
__global__ void cvt_f32_bf16(const float* __restrict__ src, bf16* __restrict__ dst, int n) {
  int i = blockIdx.x * blockDim.x + threadIdx.x;
  int stride = gridDim.x * blockDim.x;
  for (; i < n; i += stride) dst[i] = (bf16)src[i];
}

// ---------------- CDNA5 async global->LDS copy (16B per lane) ----------------
DEVINL void async_copy16(bf16* lds_dst, const bf16* gsrc) {
  unsigned ldsoff = (unsigned)(size_t)lds_dst;  // LDS aperture: addr[31:0] is the LDS offset
  unsigned long long ga = (unsigned long long)(size_t)gsrc;
  asm volatile("global_load_async_to_lds_b128 %0, %1, off"
               :: "v"(ldsoff), "v"(ga)
               : "memory");
}
DEVINL void wait_async0() { asm volatile("s_wait_asynccnt 0" ::: "memory"); }

// 64x32 bf16 row-major tile: global -> LDS via async DMA (2 x 16B per thread)
DEVINL void stage64x32_async(bf16* __restrict__ lds, const bf16* __restrict__ g,
                             int row_stride, int tid) {
#pragma unroll
  for (int i = 0; i < 2; ++i) {
    int chunk = tid + i * 128;            // 256 chunks of 8 halves
    int row = chunk >> 2;
    int c8 = (chunk & 3) * 8;
    async_copy16(lds + row * LDP + c8, g + (size_t)row * row_stride + c8);
  }
}

// ---------------- WMMA fragment loads (per ISA 16-bit layouts) ----------------
DEVINL v16bf load_a_frag(const bf16* __restrict__ Arows, int lane) {
  int m = lane & 15;
  int kh = (lane >> 4) * 8;                 // lanes 16-31 hold K+8
  const bf16* p = Arows + m * LDP + kh;
  bf16x8 lo = *(const bf16x8*)p;            // K = kh .. kh+7
  bf16x8 hi = *(const bf16x8*)(p + 16);     // K = 16+kh .. 16+kh+7
  v16bf a;
#pragma unroll
  for (int i = 0; i < 8; ++i) { a[i] = lo[i]; a[i + 8] = hi[i]; }
  return a;
}

DEVINL v16bf load_b_frag(const bf16* __restrict__ Bn, int n0, int lane) {
  int n = n0 + (lane & 15);
  int kb = (lane >> 4) * 8;                 // lanes 16-31 hold K+8
  const bf16* p = Bn + n * LDP + kb;
  bf16x8 lo = *(const bf16x8*)p;
  bf16x8 hi = *(const bf16x8*)(p + 16);
  v16bf b;
#pragma unroll
  for (int i = 0; i < 8; ++i) { b[i] = lo[i]; b[i + 8] = hi[i]; }
  return b;
}

// one 16x64 strip of D per wave: 4 WMMAs sharing one A fragment
DEVINL void wmma_tile(const bf16* __restrict__ As, const bf16* __restrict__ Bs,
                      int wave, int lane, v8f acc[4]) {
  v16bf a = load_a_frag(As + wave * 16 * LDP, lane);
#pragma unroll
  for (int j = 0; j < 4; ++j) {
    v16bf b = load_b_frag(Bs, j * 16, lane);
    acc[j] = __builtin_amdgcn_wmma_f32_16x16x32_bf16(
        false, a, false, b, (short)0, acc[j], false, false);
  }
}

// ---------------- QKV projection: Y = Xbf @ W^T (bf16 out) ----------------
__global__ __launch_bounds__(128) void qkv_gemm(const bf16* __restrict__ xbf,
                                                const bf16* __restrict__ wbf,
                                                bf16* __restrict__ qkv) {
  __shared__ alignas(16) bf16 As[2][64 * LDP];
  __shared__ alignas(16) bf16 Bs[2][64 * LDP];
  const int tid = threadIdx.x, wave = tid >> 5, lane = tid & 31;
  const int m0 = blockIdx.x * 64, n0 = blockIdx.y * 64, wsel = blockIdx.z;
  const bf16* W = wbf + (size_t)wsel * CC * CC;
  bf16* Y = qkv + (size_t)wsel * MT * CC;
  const bf16* Abase = xbf + (size_t)m0 * CC;
  const bf16* Bbase = W + (size_t)n0 * CC;   // B[c][o] = W[o][c]

  v8f acc[4] = {};
  const int nsteps = CC / 32;
  stage64x32_async(As[0], Abase, CC, tid);
  stage64x32_async(Bs[0], Bbase, CC, tid);
  wait_async0();
  __syncthreads();
  for (int s = 0; s < nsteps; ++s) {
    const int cur = s & 1, nxt = cur ^ 1;
    if (s + 1 < nsteps) {
      stage64x32_async(As[nxt], Abase + (s + 1) * 32, CC, tid);
      stage64x32_async(Bs[nxt], Bbase + (s + 1) * 32, CC, tid);
    }
    wmma_tile(As[cur], Bs[cur], wave, lane, acc);   // overlaps async DMA
    wait_async0();
    __syncthreads();
  }
  const int col = lane & 15, rh = lane >> 4;
#pragma unroll
  for (int j = 0; j < 4; ++j)
#pragma unroll
    for (int r = 0; r < 8; ++r) {
      int row = m0 + wave * 16 + r + 8 * rh;
      Y[(size_t)row * CC + n0 + j * 16 + col] = (bf16)acc[j][r];
    }
}

// ---------------- scores: S = scale * q@k^T, causal mask, online row stats ----------------
__global__ __launch_bounds__(128) void attn_scores(const bf16* __restrict__ qbf,
                                                   const bf16* __restrict__ kbf,
                                                   float* __restrict__ S,
                                                   float* __restrict__ Mrow,
                                                   float* __restrict__ Lrow) {
  __shared__ alignas(16) bf16 Qs[2][64 * LDP];
  __shared__ alignas(16) bf16 Ks[2][64 * LDP];
  const int tid = threadIdx.x, wave = tid >> 5, lane = tid & 31;
  const int nqt = TT / 64;
  const int b = blockIdx.x / nqt, qt = blockIdx.x % nqt, q0 = qt * 64;
  const bf16* qb = qbf + (size_t)b * TT * CC + (size_t)q0 * CC;
  const bf16* kb = kbf + (size_t)b * TT * CC;
  float* Sb = S + (size_t)b * TT * TT;
  const int col = lane & 15, rh = lane >> 4;
  const float scale = 0.03125f;             // 1024^-0.5
  float mrun[8], lrun[8];
#pragma unroll
  for (int r = 0; r < 8; ++r) { mrun[r] = -3.0e38f; lrun[r] = 0.f; }

  // flattened software pipeline over (key tile st, K chunk kc)
  const int total = (qt + 1) * (CC / 32);
  stage64x32_async(Qs[0], qb, CC, tid);
  stage64x32_async(Ks[0], kb, CC, tid);
  wait_async0();
  __syncthreads();

  v8f acc[4] = {};
  for (int it = 0; it < total; ++it) {
    const int cur = it & 1, nxt = cur ^ 1;
    if (it + 1 < total) {
      const int nst = (it + 1) >> 5;
      const int nkc = ((it + 1) & 31) * 32;
      stage64x32_async(Qs[nxt], qb + nkc, CC, tid);
      stage64x32_async(Ks[nxt], kb + (size_t)(nst * 64) * CC + nkc, CC, tid);
    }
    wmma_tile(Qs[cur], Ks[cur], wave, lane, acc);

    if ((it & 31) == 31) {                  // finished one 64x64 score tile
      const int s0 = (it >> 5) * 64;
      float sv[4][8];
#pragma unroll
      for (int j = 0; j < 4; ++j)
#pragma unroll
        for (int r = 0; r < 8; ++r) {
          int qrow = q0 + wave * 16 + r + 8 * rh;
          int kcol = s0 + j * 16 + col;
          float v = acc[j][r] * scale;
          if (kcol > qrow) v = -1.0e30f;    // causal mask
          sv[j][r] = v;
          Sb[(size_t)qrow * TT + kcol] = v;
        }
#pragma unroll
      for (int r = 0; r < 8; ++r) {
        float tmax = fmaxf(fmaxf(sv[0][r], sv[1][r]), fmaxf(sv[2][r], sv[3][r]));
#pragma unroll
        for (int d = 1; d < 16; d <<= 1) tmax = fmaxf(tmax, __shfl_xor(tmax, d, 32));
        float mnew = fmaxf(mrun[r], tmax);
        float ps = __expf(sv[0][r] - mnew) + __expf(sv[1][r] - mnew) +
                   __expf(sv[2][r] - mnew) + __expf(sv[3][r] - mnew);
#pragma unroll
        for (int d = 1; d < 16; d <<= 1) ps += __shfl_xor(ps, d, 32);
        lrun[r] = lrun[r] * __expf(mrun[r] - mnew) + ps;
        mrun[r] = mnew;
      }
#pragma unroll
      for (int j = 0; j < 4; ++j) acc[j] = v8f{};
    }
    wait_async0();
    __syncthreads();
  }
  if (col == 0) {
#pragma unroll
    for (int r = 0; r < 8; ++r) {
      int idx = b * TT + q0 + wave * 16 + r + 8 * rh;
      Mrow[idx] = mrun[r];
      Lrow[idx] = lrun[r];
    }
  }
}

// ---------------- out = softmax(S) @ V (P built on the fly in bf16) ----------------
__global__ __launch_bounds__(128) void attn_pv(const float* __restrict__ S,
                                               const float* __restrict__ Mrow,
                                               const float* __restrict__ Lrow,
                                               const bf16* __restrict__ vbf,
                                               float* __restrict__ out) {
  __shared__ alignas(16) bf16 Ps[2][64 * LDP];
  __shared__ alignas(16) bf16 Vs[2][64 * LDP];
  const int tid = threadIdx.x, wave = tid >> 5, lane = tid & 31;
  const int m0 = blockIdx.x * 64, n0 = blockIdx.y * 64, b = blockIdx.z;
  const float* Sb = S + (size_t)b * TT * TT;
  const bf16* vb = vbf + (size_t)b * TT * CC;
  float* ob = out + (size_t)b * TT * CC;
  const int srow = tid >> 1;                // staging: each thread owns half a P row
  const int koff = (tid & 1) * 16;
  const float mval = Mrow[b * TT + m0 + srow];
  const float linv = 1.0f / Lrow[b * TT + m0 + srow];
  const int nsteps = (m0 + 64) / 32;        // causal K limit

  // staging helpers (VALU transform / transpose, so synchronous)
  auto stage_p = [&](bf16* dst, int kc) {
    const f32x4* sp4 = (const f32x4*)(Sb + (size_t)(m0 + srow) * TT + kc + koff);
#pragma unroll
    for (int g = 0; g < 2; ++g) {
      bf16x8 pk;
#pragma unroll
      for (int q = 0; q < 2; ++q) {
        f32x4 s4 = sp4[g * 2 + q];
#pragma unroll
        for (int i = 0; i < 4; ++i)
          pk[q * 4 + i] = (bf16)(__expf(s4[i] - mval) * linv);
      }
      *(bf16x8*)(dst + srow * LDP + koff + g * 8) = pk;
    }
  };
  auto stage_v = [&](bf16* dst, int kc) {
#pragma unroll
    for (int u = 0; u < 2; ++u) {
      int unit = tid + u * 128;             // 256 units: 32 s-rows x 8 chunks
      int kk = unit >> 3;
      int c0 = (unit & 7) * 8;
      bf16x8 vv = *(const bf16x8*)(vb + (size_t)(kc + kk) * CC + n0 + c0);
#pragma unroll
      for (int jj = 0; jj < 8; ++jj) dst[(c0 + jj) * LDP + kk] = vv[jj];  // Vs[c][s]
    }
  };

  v8f acc[4] = {};
  stage_p(Ps[0], 0);
  stage_v(Vs[0], 0);
  __syncthreads();
  for (int s = 0; s < nsteps; ++s) {
    const int cur = s & 1, nxt = cur ^ 1;
    if (s + 1 < nsteps) {
      stage_p(Ps[nxt], (s + 1) * 32);
      stage_v(Vs[nxt], (s + 1) * 32);
    }
    wmma_tile(Ps[cur], Vs[cur], wave, lane, acc);
    __syncthreads();
  }
  const int col = lane & 15, rh = lane >> 4;
#pragma unroll
  for (int j = 0; j < 4; ++j)
#pragma unroll
    for (int r = 0; r < 8; ++r) {
      int row = m0 + wave * 16 + r + 8 * rh;
      ob[(size_t)row * CC + n0 + j * 16 + col] = acc[j][r];
    }
}

// ---------------- host launcher ----------------
extern "C" void kernel_launch(void* const* d_in, const int* in_sizes, int n_in,
                              void* d_out, int out_size, void* d_ws, size_t ws_size,
                              hipStream_t stream) {
  // setup_inputs order: x, Wk, Wq, Wv
  const float* x  = (const float*)d_in[0];
  const float* Wk = (const float*)d_in[1];
  const float* Wq = (const float*)d_in[2];
  const float* Wv = (const float*)d_in[3];
  float* out = (float*)d_out;

  char* ws = (char*)d_ws;
  size_t off = 0;
  auto carve = [&](size_t bytes) {
    void* p = ws + off;
    off = (off + bytes + 255) & ~(size_t)255;
    return p;
  };
  bf16*  xbf = (bf16*)carve((size_t)MT * CC * 2);
  bf16*  wbf = (bf16*)carve((size_t)3 * CC * CC * 2);   // stacked q,k,v weights
  bf16*  qkv = (bf16*)carve((size_t)3 * MT * CC * 2);   // stacked q,k,v activations
  float* S   = (float*)carve((size_t)BB * TT * TT * 4);
  float* Mr  = (float*)carve((size_t)MT * 4);
  float* Lr  = (float*)carve((size_t)MT * 4);

  // converts
  cvt_f32_bf16<<<(MT * CC + 255) / 256, 256, 0, stream>>>(x, xbf, MT * CC);
  cvt_f32_bf16<<<(CC * CC + 255) / 256, 256, 0, stream>>>(Wq, wbf + 0 * (size_t)CC * CC, CC * CC);
  cvt_f32_bf16<<<(CC * CC + 255) / 256, 256, 0, stream>>>(Wk, wbf + 1 * (size_t)CC * CC, CC * CC);
  cvt_f32_bf16<<<(CC * CC + 255) / 256, 256, 0, stream>>>(Wv, wbf + 2 * (size_t)CC * CC, CC * CC);

  // q,k,v projections (z selects weight)
  qkv_gemm<<<dim3(MT / 64, CC / 64, 3), 128, 0, stream>>>(xbf, wbf, qkv);

  // masked scores + online row stats
  attn_scores<<<dim3(BB * TT / 64), 128, 0, stream>>>(
      qkv, qkv + (size_t)MT * CC, S, Mr, Lr);

  // P @ V
  attn_pv<<<dim3(TT / 64, CC / 64, BB), 128, 0, stream>>>(
      S, Mr, Lr, qkv + 2 * (size_t)MT * CC, out);
}
